// FraudGNN_60009283060024
// MI455X (gfx1250) — compile-verified, matrix operand order checked by reading, not run on hardware
//
#include <hip/hip_runtime.h>
#include <stdint.h>

namespace {

constexpr int NU = 200000;
constexpr int NM = 50000;
constexpr int NE = 600000;
constexpr int HD = 128;
constexpr int K3 = 384;
constexpr int WSTR  = HD + 8;   // padded LDS row stride (u16) for 128-K weights (272 B)
constexpr int C1STR = K3 + 8;   // padded LDS row stride (u16) for 384-K weights (784 B)

typedef unsigned short u16;
typedef unsigned int   u32;
typedef __attribute__((ext_vector_type(16))) __bf16 v16bf;
typedef __attribute__((ext_vector_type(8)))  float  v8f;

union ABreg { v16bf v; u32 u[8]; uint4 q[2]; };

// Fragment = two contiguous 16B runs: k in [base, base+8) and [base+16, base+24).
// Matches the ISA 16-bit A/B VGPR layout with base = kc + (lane<16 ? 0 : 8).
__device__ __forceinline__ void load_frag(ABreg& r, const u16* p) {
  r.q[0] = *(const uint4*)(p);
  r.q[1] = *(const uint4*)(p + 16);
}
__device__ __forceinline__ void load_frag_lds(ABreg& r, const u16* p) {
  r.q[0] = *(const uint4*)(p);
  r.q[1] = *(const uint4*)(p + 16);
}

__device__ __forceinline__ u16 f2bf(float f) {
  union { float f; u32 u; } c; c.f = f;
  u32 u = c.u + 0x7FFFu + ((c.u >> 16) & 1u);   // round-to-nearest-even
  return (u16)(u >> 16);
}
__device__ __forceinline__ float bf2f(u16 b) {
  union { u32 u; float f; } c; c.u = ((u32)b) << 16;
  return c.f;
}

__global__ void zero_f32(float* __restrict__ p, size_t n) {
  size_t i = (size_t)blockIdx.x * blockDim.x + threadIdx.x;
  if (i < n) p[i] = 0.f;
}

// Wt[n*K + k] = bf16(W[k*N + n])   (transpose + fp32->bf16)
__global__ void wprep(const float* __restrict__ W, u16* __restrict__ Wt, int K, int N) {
  int i = blockIdx.x * blockDim.x + threadIdx.x;
  if (i >= K * N) return;
  int k = i / N, n = i % N;
  Wt[(size_t)n * K + k] = f2bf(W[i]);
}

// out[r,c] = bf16(relu(sum_k x[r,k]*W[k,c] + b[c]))  (tiny K = 2 or 3)
__global__ void node_init(const float* __restrict__ x, const float* __restrict__ W,
                          const float* __restrict__ b, int K, int n, u16* __restrict__ out) {
  size_t i = (size_t)blockIdx.x * blockDim.x + threadIdx.x;
  if (i >= (size_t)n * HD) return;
  int r = (int)(i / HD), c = (int)(i % HD);
  float acc = b[c];
  for (int k = 0; k < K; ++k) acc += x[(size_t)r * K + k] * W[(size_t)k * HD + c];
  out[i] = f2bf(acc > 0.f ? acc : 0.f);
}

__global__ void degrees(const int* __restrict__ src, const int* __restrict__ dst,
                        float* __restrict__ degU, float* __restrict__ degM) {
  int e = blockIdx.x * blockDim.x + threadIdx.x;
  if (e >= NE) return;
  atomicAdd(&degM[dst[e]], 1.f);
  atomicAdd(&degU[src[e]], 1.f);
}

// both-direction segment-sum; one thread handles 2 columns (one dword gather each way)
__global__ void scatter_sum(const u16* __restrict__ hu, const u16* __restrict__ hm,
                            const int* __restrict__ src, const int* __restrict__ dst,
                            float* __restrict__ aggM, float* __restrict__ aggU) {
  size_t t = (size_t)blockIdx.x * blockDim.x + threadIdx.x;
  if (t >= (size_t)NE * (HD / 2)) return;
  int e = (int)(t / (HD / 2));
  int c = (int)(t % (HD / 2)) * 2;
  int s = src[e], d = dst[e];
  u32 pu = *(const u32*)(hu + (size_t)s * HD + c);
  u32 pm = *(const u32*)(hm + (size_t)d * HD + c);
  atomicAdd(&aggM[(size_t)d * HD + c],     bf2f((u16)(pu & 0xFFFFu)));
  atomicAdd(&aggM[(size_t)d * HD + c + 1], bf2f((u16)(pu >> 16)));
  atomicAdd(&aggU[(size_t)s * HD + c],     bf2f((u16)(pm & 0xFFFFu)));
  atomicAdd(&aggU[(size_t)s * HD + c + 1], bf2f((u16)(pm >> 16)));
}

__global__ void mean_cvt(const float* __restrict__ agg, const float* __restrict__ deg,
                         int n, u16* __restrict__ out) {
  size_t i = (size_t)blockIdx.x * blockDim.x + threadIdx.x;
  if (i >= (size_t)n * HD) return;
  int r = (int)(i / HD);
  out[i] = f2bf(agg[i] / fmaxf(deg[r], 1.f));
}

// new = relu(Aagg @ Wl + bias + Ah @ Wr) via v_wmma_f32_16x16x32_bf16.
// One wave computes a 16-row x 128-col strip. Both weight matrices (pre-transposed
// [N=128][K=128] bf16) are staged once per block into padded LDS; B fragments are
// then conflict-free ds_load_b128s shared by all 8 waves.
__global__ __launch_bounds__(256) void sage_gemm(
    const u16* __restrict__ Aagg, const u16* __restrict__ Ah,
    const u16* __restrict__ Wlt,  const u16* __restrict__ Wrt,
    const float* __restrict__ bias, u16* __restrict__ out, int ntiles) {
  extern __shared__ u16 lds_w[];   // 2 * HD * WSTR u16 = 69,632 B

  {
    const uint4* s0 = (const uint4*)Wlt;   // HD rows x 16 uint4
    const uint4* s1 = (const uint4*)Wrt;
    for (int i = threadIdx.x; i < HD * 16; i += 256) {
      int r = i >> 4, q = i & 15;
      *(uint4*)(&lds_w[(size_t)r * WSTR + q * 8]) = s0[i];
      *(uint4*)(&lds_w[(size_t)(HD + r) * WSTR + q * 8]) = s1[i];
    }
  }
  __syncthreads();

  int w = blockIdx.x * 8 + (threadIdx.x >> 5);
  if (w >= ntiles) return;
  int lane = threadIdx.x & 31;
  int lm   = lane & 15;
  int koff = (lane < 16) ? 0 : 8;        // ISA 16-bit A/B lane->K mapping
  int r0   = w * 16;
  int row  = r0 + lm;

  v8f acc[8];
#pragma unroll
  for (int nt = 0; nt < 8; ++nt)
#pragma unroll
    for (int i = 0; i < 8; ++i) acc[nt][i] = 0.f;

#pragma unroll
  for (int pass = 0; pass < 2; ++pass) {
    const u16* Arow = (pass ? Ah : Aagg) + (size_t)row * HD + koff;
    const u16* Wl   = &lds_w[(size_t)pass * HD * WSTR + lm * WSTR + koff];
#pragma unroll
    for (int kc = 0; kc < HD; kc += 32) {
      ABreg a;
      load_frag(a, Arow + kc);
#pragma unroll
      for (int nt = 0; nt < 8; ++nt) {
        ABreg b;
        load_frag_lds(b, Wl + (size_t)nt * 16 * WSTR + kc);
        acc[nt] = __builtin_amdgcn_wmma_f32_16x16x32_bf16(
            false, a.v, false, b.v, (short)0, acc[nt], false, false);
      }
    }
  }

  int rbase = r0 + ((lane < 16) ? 0 : 8);   // C/D layout: VGPR i -> M = i (+8 for hi half)
#pragma unroll
  for (int nt = 0; nt < 8; ++nt) {
    int col = nt * 16 + lm;
    float bv = bias[col];
#pragma unroll
    for (int i = 0; i < 8; ++i) {
      float v = acc[nt][i] + bv;
      out[(size_t)(rbase + i) * HD + col] = f2bf(v > 0.f ? v : 0.f);
    }
  }
}

// Fused per-edge MLP: hid = relu([h_u[src], h_m[dst], e] @ c1 + c1b); out = hid @ c2 + c2b
// One wave = 16 edges; K = 384 in 12 chunks of 32; 96 WMMAs; shfl_xor column reduction.
// c1 (pre-transposed [128][384] bf16) staged once per block into padded LDS.
__global__ __launch_bounds__(256) void edge_mlp(
    const u16* __restrict__ hu, const u16* __restrict__ hm, const u16* __restrict__ ee,
    const u16* __restrict__ c1t, const float* __restrict__ c1b,
    const float* __restrict__ c2w, const float* __restrict__ c2b,
    const int* __restrict__ src, const int* __restrict__ dst,
    float* __restrict__ out) {
  extern __shared__ u16 lds_c1[];   // HD * C1STR u16 = 100,352 B

  {
    const uint4* s = (const uint4*)c1t;    // HD rows x 24 uint4
    for (int i = threadIdx.x; i < HD * 24; i += 256) {
      int r = i / 24, q = i % 24;
      *(uint4*)(&lds_c1[(size_t)r * C1STR + q * 8]) = s[i];
    }
  }
  __syncthreads();

  int w = blockIdx.x * 8 + (threadIdx.x >> 5);
  if (w >= NE / 16) return;
  int lane = threadIdx.x & 31;
  int lm   = lane & 15;
  int koff = (lane < 16) ? 0 : 8;
  int e0   = w * 16;
  int erow = e0 + lm;

  const u16* rowU = hu + (size_t)src[erow] * HD + koff;
  const u16* rowM = hm + (size_t)dst[erow] * HD + koff;
  const u16* rowE = ee + (size_t)erow * HD + koff;
  const u16* Wc   = &lds_c1[(size_t)lm * C1STR + koff];

  v8f acc[8];
#pragma unroll
  for (int nt = 0; nt < 8; ++nt)
#pragma unroll
    for (int i = 0; i < 8; ++i) acc[nt][i] = 0.f;

#pragma unroll
  for (int kc = 0; kc < K3; kc += 32) {
    const u16* Arow;
    if (kc < HD)          Arow = rowU + kc;
    else if (kc < 2 * HD) Arow = rowM + (kc - HD);
    else                  Arow = rowE + (kc - 2 * HD);
    ABreg a;
    load_frag(a, Arow);
#pragma unroll
    for (int nt = 0; nt < 8; ++nt) {
      ABreg b;
      load_frag_lds(b, Wc + (size_t)nt * 16 * C1STR + kc);
      acc[nt] = __builtin_amdgcn_wmma_f32_16x16x32_bf16(
          false, a.v, false, b.v, (short)0, acc[nt], false, false);
    }
  }

  float s[8];
#pragma unroll
  for (int i = 0; i < 8; ++i) s[i] = 0.f;
#pragma unroll
  for (int nt = 0; nt < 8; ++nt) {
    int col = nt * 16 + lm;
    float b1 = c1b[col];
    float w2 = c2w[col];
#pragma unroll
    for (int i = 0; i < 8; ++i) {
      float h = acc[nt][i] + b1;
      h = h > 0.f ? h : 0.f;
      s[i] += h * w2;
    }
  }
  // reduce over the 16 lanes of each half (columns); xor masks 1,2,4,8 stay in-half
#pragma unroll
  for (int m = 8; m >= 1; m >>= 1)
#pragma unroll
    for (int i = 0; i < 8; ++i) s[i] += __shfl_xor(s[i], m, 32);

  if (lm == 0) {
    float cb = c2b[0];
    int rb = e0 + ((lane < 16) ? 0 : 8);
#pragma unroll
    for (int i = 0; i < 8; ++i) out[rb + i] = s[i] + cb;
  }
}

} // namespace

extern "C" void kernel_launch(void* const* d_in, const int* in_sizes, int n_in,
                              void* d_out, int out_size, void* d_ws, size_t ws_size,
                              hipStream_t stream) {
  (void)in_sizes; (void)n_in; (void)out_size; (void)ws_size;

  const float* user_x  = (const float*)d_in[0];
  const float* merch_x = (const float*)d_in[1];
  const float* edge_at = (const float*)d_in[2];
  const float* u_lw = (const float*)d_in[3];
  const float* u_lb = (const float*)d_in[4];
  const float* m_lw = (const float*)d_in[5];
  const float* m_lb = (const float*)d_in[6];
  const float* e_lw = (const float*)d_in[7];
  const float* e_lb = (const float*)d_in[8];
  const float* u2m_wl = (const float*)d_in[9];
  const float* u2m_bl = (const float*)d_in[10];
  const float* u2m_wr = (const float*)d_in[11];
  const float* m2u_wl = (const float*)d_in[12];
  const float* m2u_bl = (const float*)d_in[13];
  const float* m2u_wr = (const float*)d_in[14];
  const float* c1w = (const float*)d_in[15];
  const float* c1b = (const float*)d_in[16];
  const float* c2w = (const float*)d_in[17];
  const float* c2b = (const float*)d_in[18];
  const int*   eidx = (const int*)d_in[19];
  const int* src = eidx;
  const int* dst = eidx + NE;
  float* out = (float*)d_out;

  char* base = (char*)d_ws;
  size_t off = 0;
  auto alloc = [&](size_t bytes) -> void* {
    void* p = base + off;
    off += (bytes + 255) & ~(size_t)255;
    return p;
  };
  u16*  huA   = (u16*)alloc((size_t)NU * HD * 2);
  u16*  huB   = (u16*)alloc((size_t)NU * HD * 2);
  u16*  hmA   = (u16*)alloc((size_t)NM * HD * 2);
  u16*  hmB   = (u16*)alloc((size_t)NM * HD * 2);
  u16*  ebf   = (u16*)alloc((size_t)NE * HD * 2);
  float* aggU = (float*)alloc((size_t)NU * HD * 4);
  float* aggM = (float*)alloc((size_t)NM * HD * 4);
  u16*  aggUb = (u16*)alloc((size_t)NU * HD * 2);
  u16*  aggMb = (u16*)alloc((size_t)NM * HD * 2);
  float* degU = (float*)alloc((size_t)NU * 4);
  float* degM = (float*)alloc((size_t)NM * 4);
  u16*  wU2Ml = (u16*)alloc((size_t)2 * HD * HD * 2);
  u16*  wU2Mr = (u16*)alloc((size_t)2 * HD * HD * 2);
  u16*  wM2Ul = (u16*)alloc((size_t)2 * HD * HD * 2);
  u16*  wM2Ur = (u16*)alloc((size_t)2 * HD * HD * 2);
  u16*  c1t   = (u16*)alloc((size_t)HD * K3 * 2);

  dim3 B(256);
  auto g1 = [](size_t n) { return dim3((unsigned)((n + 255) / 256)); };
  const int HH = HD * HD;
  const size_t sage_lds = (size_t)2 * HD * WSTR * sizeof(u16);   // 69,632 B
  const size_t edge_lds = (size_t)HD * C1STR * sizeof(u16);      // 100,352 B

  // --- weight prep (transpose + bf16) ---
  for (int l = 0; l < 2; ++l) {
    wprep<<<g1(HH), B, 0, stream>>>(u2m_wl + (size_t)l * HH, wU2Ml + (size_t)l * HH, HD, HD);
    wprep<<<g1(HH), B, 0, stream>>>(u2m_wr + (size_t)l * HH, wU2Mr + (size_t)l * HH, HD, HD);
    wprep<<<g1(HH), B, 0, stream>>>(m2u_wl + (size_t)l * HH, wM2Ul + (size_t)l * HH, HD, HD);
    wprep<<<g1(HH), B, 0, stream>>>(m2u_wr + (size_t)l * HH, wM2Ur + (size_t)l * HH, HD, HD);
  }
  wprep<<<g1((size_t)K3 * HD), B, 0, stream>>>(c1w, c1t, K3, HD);

  // --- input projections (bf16 hidden states) ---
  node_init<<<g1((size_t)NU * HD), B, 0, stream>>>(user_x,  u_lw, u_lb, 3, NU, huA);
  node_init<<<g1((size_t)NM * HD), B, 0, stream>>>(merch_x, m_lw, m_lb, 2, NM, hmA);
  node_init<<<g1((size_t)NE * HD), B, 0, stream>>>(edge_at, e_lw, e_lb, 2, NE, ebf);

  // --- degrees (invariant across layers) ---
  zero_f32<<<g1(NU), B, 0, stream>>>(degU, NU);
  zero_f32<<<g1(NM), B, 0, stream>>>(degM, NM);
  degrees<<<g1(NE), B, 0, stream>>>(src, dst, degU, degM);

  // --- 2 SAGE layers ---
  u16 *huC = huA, *huN = huB, *hmC = hmA, *hmN = hmB;
  for (int l = 0; l < 2; ++l) {
    zero_f32<<<g1((size_t)NU * HD), B, 0, stream>>>(aggU, (size_t)NU * HD);
    zero_f32<<<g1((size_t)NM * HD), B, 0, stream>>>(aggM, (size_t)NM * HD);
    scatter_sum<<<g1((size_t)NE * (HD / 2)), B, 0, stream>>>(huC, hmC, src, dst, aggM, aggU);
    mean_cvt<<<g1((size_t)NM * HD), B, 0, stream>>>(aggM, degM, NM, aggMb);
    mean_cvt<<<g1((size_t)NU * HD), B, 0, stream>>>(aggU, degU, NU, aggUb);
    sage_gemm<<<dim3((NM / 16 + 7) / 8), B, sage_lds, stream>>>(
        aggMb, hmC, wU2Ml + (size_t)l * HH, wU2Mr + (size_t)l * HH,
        u2m_bl + (size_t)l * HD, hmN, NM / 16);
    sage_gemm<<<dim3((NU / 16 + 7) / 8), B, sage_lds, stream>>>(
        aggUb, huC, wM2Ul + (size_t)l * HH, wM2Ur + (size_t)l * HH,
        m2u_bl + (size_t)l * HD, huN, NU / 16);
    u16* t;
    t = huC; huC = huN; huN = t;
    t = hmC; hmC = hmN; hmN = t;
  }

  // --- fused per-edge MLP ---
  edge_mlp<<<dim3((NE / 16 + 7) / 8), B, edge_lds, stream>>>(
      huC, hmC, ebf, c1t, c1b, c2w, c2b, src, dst, out);
}
